// Attention_82841329205539
// MI455X (gfx1250) — matrix operand order repaired and checked by reading, not compiled, not run on hardware
//
#include <hip/hip_runtime.h>
#include <hip/hip_bf16.h>

// ---------------------------------------------------------------------------
// Sliding-window attention, MI455X (gfx1250, wave32, WMMA).
// All GEMM work on v_wmma_f32_16x16x32_bf16 (fp32 accumulate).
// Round 2: software-pipelined GEMM (32x64 per wave, double-buffered
// fragments) so WMMAs overlap in-flight global loads instead of stalling
// on s_wait_loadcnt 0.
// ---------------------------------------------------------------------------

typedef __bf16 bf16_t;
typedef __attribute__((ext_vector_type(16))) __bf16 v16bf;
typedef __attribute__((ext_vector_type(8)))  __bf16 v8bf;
typedef __attribute__((ext_vector_type(8)))  float  v8f;

static constexpr int Bb   = 2;
static constexpr int Nn   = 2048;
static constexpr int Cc   = 768;
static constexpr int Hh   = 12;
static constexpr int Dd   = 64;      // head dim
static constexpr int C3   = 3 * Cc;  // 2304
static constexpr int Mm   = Bb * Nn; // 4096 rows of token matrix
static constexpr float SCALE = 0.125f;   // D^-0.5 = 1/8
static constexpr int NT   = 9;       // key tiles covering [m0-64, m0+79]
static constexpr int KPAD = 160;     // padded P width (9*16=144 -> 160)

// --- fragment loaders (wave32; this code's contiguous k order k = 16*(lane/16) + s, same in A and B) ----
// A (16-bit, 16x32): lane<16 holds row M=lane, K = kbase+0..15 contiguous;
//                    lane>=16 same row, K = kbase+16..31 (matches B)
__device__ __forceinline__ v16bf load_a_frag(const bf16_t* __restrict__ row,
                                             int kbase, int lane) {
  return *(const v16bf*)(row + kbase + ((lane >> 4) << 4));
}

// B (16-bit, 32x16): lane<16 holds column N=lane, K = kbase+0..15 contiguous;
//                    lane>=16 same column, K = kbase+16..31.
__device__ __forceinline__ v16bf load_b_frag(const bf16_t* __restrict__ col,
                                             int kbase, int lane) {
  return *(const v16bf*)(col + kbase + ((lane >> 4) << 4));
}

__device__ __forceinline__ v8f wmma_bf16(v16bf a, v16bf b, v8f c) {
  return __builtin_amdgcn_wmma_f32_16x16x32_bf16(false, a, false, b,
                                                 (short)0, c, false, false);
}

// ---------------------------------------------------------------------------
// fp32 -> bf16 conversion (one-time; halves operand bytes into the GEMMs)
// ---------------------------------------------------------------------------
__global__ void cvt_f32_bf16(const float* __restrict__ in,
                             bf16_t* __restrict__ out, int n) {
  int i = blockIdx.x * blockDim.x + threadIdx.x;
  if (i < n) out[i] = (bf16_t)in[i];
}

// ---------------------------------------------------------------------------
// C = A[M,K](bf16) * W[N,K]^T (bf16)  -> fp32 (+bias) and/or bf16 output.
// Wave computes a 32x64 tile: 2 A-frags x 4 B-frags = 8 WMMAs per K-step,
// double-buffered so step k's WMMAs overlap step k+32's loads.
// ---------------------------------------------------------------------------
__global__ __launch_bounds__(256) void gemm_bf16_nt(
    const bf16_t* __restrict__ A, const bf16_t* __restrict__ W,
    const float* __restrict__ bias, float* __restrict__ outf,
    bf16_t* __restrict__ outb, int M, int Nout, int K) {
  const int lane = threadIdx.x & 31;
  const int wave = threadIdx.x >> 5;
  const int l15 = lane & 15;
  const int m0 = (blockIdx.y * 8 + wave) * 32;
  const int n0 = blockIdx.x * 64;
  if (m0 >= M) return;

  const bf16_t* arow0 = A + (size_t)(m0 + l15) * K;
  const bf16_t* arow1 = A + (size_t)(m0 + 16 + l15) * K;
  const bf16_t* wrow0 = W + (size_t)(n0 +  0 + l15) * K;
  const bf16_t* wrow1 = W + (size_t)(n0 + 16 + l15) * K;
  const bf16_t* wrow2 = W + (size_t)(n0 + 32 + l15) * K;
  const bf16_t* wrow3 = W + (size_t)(n0 + 48 + l15) * K;

  v8f c00 = {}, c01 = {}, c02 = {}, c03 = {};
  v8f c10 = {}, c11 = {}, c12 = {}, c13 = {};

  // prologue: fragments for k = 0
  v16bf a0 = load_a_frag(arow0, 0, lane);
  v16bf a1 = load_a_frag(arow1, 0, lane);
  v16bf b0 = load_b_frag(wrow0, 0, lane);
  v16bf b1 = load_b_frag(wrow1, 0, lane);
  v16bf b2 = load_b_frag(wrow2, 0, lane);
  v16bf b3 = load_b_frag(wrow3, 0, lane);

  for (int k = 0; k < K; k += 32) {
    // issue next-step loads first (wrap on last iter: redundant but branchfree)
    const int kn = (k + 32 < K) ? (k + 32) : 0;
    __builtin_prefetch(arow0 + k + 512, 0, 3);     // global_prefetch_b8 (WGP)
    v16bf na0 = load_a_frag(arow0, kn, lane);
    v16bf na1 = load_a_frag(arow1, kn, lane);
    v16bf nb0 = load_b_frag(wrow0, kn, lane);
    v16bf nb1 = load_b_frag(wrow1, kn, lane);
    v16bf nb2 = load_b_frag(wrow2, kn, lane);
    v16bf nb3 = load_b_frag(wrow3, kn, lane);

    // math on current (already-resident) fragments
    c00 = wmma_bf16(a0, b0, c00);
    c01 = wmma_bf16(a0, b1, c01);
    c02 = wmma_bf16(a0, b2, c02);
    c03 = wmma_bf16(a0, b3, c03);
    c10 = wmma_bf16(a1, b0, c10);
    c11 = wmma_bf16(a1, b1, c11);
    c12 = wmma_bf16(a1, b2, c12);
    c13 = wmma_bf16(a1, b3, c13);

    a0 = na0; a1 = na1;
    b0 = nb0; b1 = nb1; b2 = nb2; b3 = nb3;
  }

  // C/D layout: VGPR r -> row base + 8*(lane>=16) + r, col base + (lane&15)
  v8f cs[2][4] = {{c00, c01, c02, c03}, {c10, c11, c12, c13}};
#pragma unroll
  for (int half = 0; half < 2; ++half) {
    const int rbase = m0 + half * 16 + ((lane >> 4) << 3);
#pragma unroll
    for (int j = 0; j < 4; ++j) {
      const int col = n0 + j * 16 + l15;
      const float badd = bias ? bias[col] : 0.0f;
#pragma unroll
      for (int r = 0; r < 8; ++r) {
        const size_t idx = (size_t)(rbase + r) * Nout + col;
        const float v = cs[half][j][r];
        if (outf) outf[idx] = v + badd;
        if (outb) outb[idx] = (bf16_t)v;
      }
    }
  }
}

// ---------------------------------------------------------------------------
// V transpose: qkv[(b*N+n)*3C + 2C + h*D + d] -> vt[((b*H+h)*D+d)*N + n]
// (makes P*V B-fragments contiguous per-lane loads)
// ---------------------------------------------------------------------------
__global__ void transpose_v(const bf16_t* __restrict__ qkvb,
                            bf16_t* __restrict__ vt) {
  int idx = blockIdx.x * blockDim.x + threadIdx.x;  // B*H*D*N threads
  int n = idx & (Nn - 1);
  int t = idx >> 11;      // /N
  int d = t & (Dd - 1);
  t >>= 6;                // /D
  int h = t % Hh;
  int b = t / Hh;
  vt[idx] = qkvb[(size_t)(b * Nn + n) * C3 + 2 * Cc + h * Dd + d];
}

// ---------------------------------------------------------------------------
// Sliding-window attention. One wave per (b, h, 16-query tile).
// Window |i-j| <= 64 -> keys in [m0-64, m0+79]: 9 aligned key tiles of 16.
// ---------------------------------------------------------------------------
__global__ __launch_bounds__(256) void attn_sw(
    const bf16_t* __restrict__ qkvb, const bf16_t* __restrict__ vt,
    bf16_t* __restrict__ attnb) {
  __shared__ __align__(32) bf16_t plds[8][16][KPAD];

  const int lane = threadIdx.x & 31;
  const int wave = threadIdx.x >> 5;
  const int wg = blockIdx.x * 8 + wave;      // 0 .. B*H*(N/16)-1 == 3071
  const int qt = wg & 127;                   // N/16 = 128
  const int h  = (wg >> 7) % Hh;
  const int b  = wg / (128 * Hh);
  const int m0 = qt * 16;
  const int l15 = lane & 15;
  const int lh  = lane >> 4;
  const int jb0 = m0 - 64;

  // ---- Q fragments (rows = queries, K-dim = d, contiguous in qkv row) ----
  const bf16_t* qrow = qkvb + (size_t)(b * Nn + m0 + l15) * C3 + h * Dd;
  v16bf qa0 = load_a_frag(qrow, 0, lane);
  v16bf qa1 = load_a_frag(qrow, 32, lane);

  // ---- S = (Q K^T) * SCALE, masked ----
  float s[NT][8];
#pragma unroll
  for (int t = 0; t < NT; ++t) {
    const int jbase = jb0 + t * 16;                 // aligned, fully in/out
    const bool valid = (jbase >= 0) && (jbase < Nn);
    v8f acc = {};
    if (valid) {  // wave-uniform branch: EXEC all-1s inside (WMMA-safe)
      const bf16_t* krow =
          qkvb + (size_t)(b * Nn + jbase + l15) * C3 + Cc + h * Dd;
      v16bf kb0 = load_b_frag(krow, 0, lane);
      v16bf kb1 = load_b_frag(krow, 32, lane);
      acc = wmma_bf16(qa0, kb0, acc);
      acc = wmma_bf16(qa1, kb1, acc);
    }
#pragma unroll
    for (int r = 0; r < 8; ++r) {
      const int qi = m0 + (lh << 3) + r;
      const int kj = jbase + l15;
      const int dlt = qi - kj;
      const bool inwin = valid && (dlt <= 64) && (dlt >= -64);
      s[t][r] = inwin ? acc[r] * SCALE : -1e30f;
    }
  }

  // ---- row softmax: row's 16 cols live in one 16-lane group per VGPR r ----
  float inv[8];
#pragma unroll
  for (int r = 0; r < 8; ++r) {
    float m = -1e30f;
#pragma unroll
    for (int t = 0; t < NT; ++t) m = fmaxf(m, s[t][r]);
#pragma unroll
    for (int off = 8; off > 0; off >>= 1) m = fmaxf(m, __shfl_xor(m, off));
    float sum = 0.0f;
#pragma unroll
    for (int t = 0; t < NT; ++t) {
      const float e = __expf(s[t][r] - m);  // masked: exp(-1e30) -> 0
      s[t][r] = e;
      sum += e;
    }
#pragma unroll
    for (int off = 8; off > 0; off >>= 1) sum += __shfl_xor(sum, off);
    inv[r] = 1.0f / sum;                    // window always contains diagonal
  }

  // ---- stage P (16 x 160, zero padded) in LDS to reswizzle C->A layout ----
  bf16_t* pw = &plds[wave][0][0];
  for (int i = lane; i < 16 * KPAD; i += 32) pw[i] = (bf16_t)0.0f;
  __syncthreads();
#pragma unroll
  for (int t = 0; t < NT; ++t)
#pragma unroll
    for (int r = 0; r < 8; ++r)
      plds[wave][(lh << 3) + r][t * 16 + l15] = (bf16_t)(s[t][r] * inv[r]);
  __syncthreads();

  // ---- O = P * V  (A from LDS, B from vt; K = padded key index) ----
  const bf16_t* prow = &plds[wave][l15][0];
#pragma unroll
  for (int dt = 0; dt < 4; ++dt) {
    v8f o = {};
    const bf16_t* vrow =
        vt + (size_t)((b * Hh + h) * Dd + dt * 16 + l15) * Nn;
#pragma unroll
    for (int kk = 0; kk < KPAD; kk += 32) {
      v16bf pa = load_a_frag(prow, kk, lane);
      const int j0 = jb0 + kk + (lh << 4);
      v16bf vb;
      if (j0 >= 0 && j0 + 16 <= Nn) {
        vb = *(const v16bf*)(vrow + j0);
      } else {               // boundary tiles: clamp (P is 0 there anyway)
#pragma unroll
        for (int i = 0; i < 16; ++i) {
          int jj = j0 + i;
          jj = jj < 0 ? 0 : (jj > Nn - 1 ? Nn - 1 : jj);
          vb[i] = vrow[jj];
        }
      }
      o = wmma_bf16(pa, vb, o);             // after reconvergence: EXEC full
    }
#pragma unroll
    for (int r = 0; r < 8; ++r) {
      const size_t row = (size_t)(b * Nn + m0 + (lh << 3) + r);
      attnb[row * Cc + h * Dd + dt * 16 + l15] = (bf16_t)o[r];
    }
  }
}

// ---------------------------------------------------------------------------
extern "C" void kernel_launch(void* const* d_in, const int* in_sizes, int n_in,
                              void* d_out, int out_size, void* d_ws,
                              size_t ws_size, hipStream_t stream) {
  const float* x      = (const float*)d_in[0];   // [B,N,C]
  const float* w_qkv  = (const float*)d_in[1];   // [3C,C]
  const float* w_proj = (const float*)d_in[2];   // [C,C]
  const float* b_proj = (const float*)d_in[3];   // [C]
  float* out = (float*)d_out;                    // [B,N,C]

  char* ws = (char*)d_ws;
  size_t off = 0;
  auto take = [&](size_t elems) {
    void* p = ws + off;
    off = (off + elems * sizeof(bf16_t) + 255) & ~(size_t)255;
    return (bf16_t*)p;
  };
  bf16_t* xb     = take((size_t)Mm * Cc);        // x in bf16
  bf16_t* wqkvb  = take((size_t)C3 * Cc);        // w_qkv in bf16
  bf16_t* wprojb = take((size_t)Cc * Cc);        // w_proj in bf16
  bf16_t* qkvb   = take((size_t)Mm * C3);        // qkv projection (bf16)
  bf16_t* vtb    = take((size_t)Bb * Hh * Dd * Nn); // V transposed
  bf16_t* attnb  = take((size_t)Mm * Cc);        // attention output (bf16)
  (void)ws_size; (void)in_sizes; (void)n_in; (void)out_size;

  // 1) downconvert operands to bf16
  cvt_f32_bf16<<<(Mm * Cc + 255) / 256, 256, 0, stream>>>(x, xb, Mm * Cc);
  cvt_f32_bf16<<<(C3 * Cc + 255) / 256, 256, 0, stream>>>(w_qkv, wqkvb, C3 * Cc);
  cvt_f32_bf16<<<(Cc * Cc + 255) / 256, 256, 0, stream>>>(w_proj, wprojb, Cc * Cc);

  // 2) qkv = x @ w_qkv^T   [4096 x 2304]
  {
    dim3 grid(C3 / 64, Mm / 256);   // 32 rows per wave, 8 waves per block
    gemm_bf16_nt<<<grid, 256, 0, stream>>>(xb, wqkvb, nullptr, nullptr, qkvb,
                                           Mm, C3, Cc);
  }

  // 3) V -> [B,H,D,N]
  transpose_v<<<(Bb * Hh * Dd * Nn) / 256, 256, 0, stream>>>(qkvb, vtb);

  // 4) sliding-window attention (3072 wave-tiles, 8 waves/block)
  attn_sw<<<(Bb * Hh * (Nn / 16)) / 8, 256, 0, stream>>>(qkvb, vtb, attnb);

  // 5) out = attn @ w_proj^T + b_proj   [4096 x 768] fp32
  {
    dim3 grid(Cc / 64, Mm / 256);
    gemm_bf16_nt<<<grid, 256, 0, stream>>>(attnb, wprojb, b_proj, out, nullptr,
                                           Mm, Cc, Cc);
  }
}
